// NonLocalCA_41953240547849
// MI455X (gfx1250) — matrix operand order repaired and checked by reading, not compiled, not run on hardware
//
#include <hip/hip_runtime.h>
#include <hip/hip_bf16.h>
#include <math.h>

// ---------------------------------------------------------------------------
// NonLocal (embedded-gaussian) block on 4 quadrants, MI455X / gfx1250.
//   Q/K/V projections -> f16, transposed flash-attention with
//   v_wmma_f32_16x16x32_f16 (S^T = K Q^T, Y^T = V^T P^T: softmax stats are
//   within-lane, P^T relayout done in registers), fused output projection +
//   training-mode BatchNorm + residual.
// Shapes: B=2, C=64, H=W=128 -> per (quadrant q, batch b): N=4096 tokens, Ci=32.
// ---------------------------------------------------------------------------

typedef __attribute__((ext_vector_type(16))) _Float16 v16h;
typedef __attribute__((ext_vector_type(2)))  __fp16   v2fp16;
typedef __attribute__((ext_vector_type(8)))  float    v8f;
typedef __attribute__((ext_vector_type(4)))  float    vf4;

#define NTOK 4096
#define CI   32
#define CCH  64

// Load 16 halves as two 16-byte chunks into a v16h WMMA operand register set.
__device__ __forceinline__ v16h ld16(const _Float16* p0, const _Float16* p1) {
    union { vf4 f[2]; v16h v; } u;
    u.f[0] = *(const vf4*)p0;
    u.f[1] = *(const vf4*)p1;
    return u.v;
}

// Pack two f32 into one VGPR holding two f16 (v_cvt_pk_f16_f32 rtz).
__device__ __forceinline__ float pack2h(float a, float b) {
    union { v2fp16 h2; float f; } u;
    u.h2 = __builtin_amdgcn_cvt_pkrtz(a, b);
    return u.f;
}

// ---------------------------------------------------------------------------
// Kernel 1: projections  th/ph/g : [Ci,C] x x[C, n]  (+bias)
//   Q, K stored row-major [bq][N][32] f16 ; V stored transposed [bq][32][N] f16
// grid: 4q * 2b * 64rows = 512 blocks, 256 threads
// ---------------------------------------------------------------------------
__global__ void __launch_bounds__(256)
proj_kernel(const float* __restrict__ x,
            const float* __restrict__ g_w,  const float* __restrict__ g_b,
            const float* __restrict__ th_w, const float* __restrict__ th_b,
            const float* __restrict__ ph_w, const float* __restrict__ ph_b,
            _Float16* __restrict__ Q, _Float16* __restrict__ K,
            _Float16* __restrict__ Vt)
{
    const int blk = blockIdx.x;
    const int q = blk >> 7;            // 0..3
    const int b = (blk >> 6) & 1;      // 0..1
    const int i = blk & 63;            // quadrant row 0..63
    const int qy = q >> 1, qx = q & 1;
    const int bq = q * 2 + b;
    const int tid = threadIdx.x;

    __shared__ float xs[CCH][64];        // x tile [c][j]
    __shared__ float wsm[3][CI][CCH];    // theta, phi, g weights

    for (int idx = tid; idx < CI * CCH; idx += 256) {
        wsm[0][idx >> 6][idx & 63] = th_w[idx];
        wsm[1][idx >> 6][idx & 63] = ph_w[idx];
        wsm[2][idx >> 6][idx & 63] = g_w[idx];
    }
    for (int idx = tid; idx < CCH * 64; idx += 256) {
        const int c = idx >> 6, j = idx & 63;
        xs[c][j] = x[(((size_t)b * CCH + c) * 128 + (qy * 64 + i)) * 128 + qx * 64 + j];
    }
    __syncthreads();

    for (int idx = tid; idx < 64 * CI; idx += 256) {
        const int j = idx & 63;         // pixel column within quadrant row
        const int o = idx >> 6;         // output channel 0..31
        float a0 = 0.f, a1 = 0.f, a2 = 0.f;
        #pragma unroll 8
        for (int c = 0; c < CCH; ++c) {
            const float xv = xs[c][j];
            a0 += wsm[0][o][c] * xv;
            a1 += wsm[1][o][c] * xv;
            a2 += wsm[2][o][c] * xv;
        }
        const int n = i * 64 + j;
        const size_t base = ((size_t)bq * NTOK + n) * CI + o;
        Q[base] = (_Float16)(a0 + th_b[o]);
        K[base] = (_Float16)(a1 + ph_b[o]);
        Vt[(size_t)bq * (CI * NTOK) + (size_t)o * NTOK + n] = (_Float16)(a2 + g_b[o]);
    }
}

// ---------------------------------------------------------------------------
// Kernel 2: transposed flash attention.
//   S^T = K . Q^T  (A = K-tile 16keys x 32ci, B = Q-tile 32ci x 16queries)
//   Y^T = V^T . P^T (A = Vt-tile 16ch x 32keys, B = P^T assembled in regs)
// grid: (32, 8): blockIdx.y = bq, 8 waves/block, 16 queries per wave.
// Operand layouts assumed (per ISA 7.12.2):
//   A (16x32 f16): lane(m=l&15,h=l>>4): row m, K in {8h..8h+7, 16+8h..16+8h+7}
//   B (32x16 f16): lane(n=l&15,h=l>>4): col n, K order as A's; P^T and Vt use 16h..16h+15
//   C/D (16x16 f32): VGPR r, lane l -> row r+8*(l>>4), col l&15
// ---------------------------------------------------------------------------
__global__ void __launch_bounds__(256)
flash_kernel(const _Float16* __restrict__ Q, const _Float16* __restrict__ K,
             const _Float16* __restrict__ Vt, float* __restrict__ Y)
{
    const int bq   = blockIdx.y;
    const int wave = threadIdx.x >> 5;
    const int lane = threadIdx.x & 31;
    const int qc = lane & 15;          // query column / A-row index
    const int h  = lane >> 4;
    const int qbase = (blockIdx.x * 8 + wave) * 16;

    // B operand: Q^T tile (32ci x 16 queries), persistent: ci {8h..8h+7, 16+8h..16+8h+7}
    const _Float16* Qp = Q + ((size_t)bq * NTOK + qbase + qc) * CI + 8 * h;
    const v16h qb = ld16(Qp, Qp + 16);

    float mr = -3.0e38f, lr = 0.f;     // per-query running max / sum (per lane)
    v8f acc0 = {}; v8f acc1 = {};      // Y^T tiles: ch 0-15 / 16-31

    const _Float16* Kbq = K + (size_t)bq * NTOK * CI;
    const _Float16* Vbq = Vt + (size_t)bq * (CI * NTOK);

    #pragma unroll 2
    for (int kb = 0; kb < NTOK; kb += 32) {
        // --- S^T tiles: keys kb+0..15 (s0), kb+16..31 (s1) ---
        const _Float16* k0 = Kbq + (size_t)(kb + qc) * CI;
        const _Float16* k1 = Kbq + (size_t)(kb + 16 + qc) * CI;
        const v16h ka0 = ld16(k0 + 8 * h, k0 + 16 + 8 * h);
        const v16h ka1 = ld16(k1 + 8 * h, k1 + 16 + 8 * h);
        const v8f z = {};
        v8f s0 = __builtin_amdgcn_wmma_f32_16x16x32_f16(false, ka0, false, qb,
                                                        (short)0, z, false, false);
        v8f s1 = __builtin_amdgcn_wmma_f32_16x16x32_f16(false, ka1, false, qb,
                                                        (short)0, z, false, false);
        // lane (qc,h) holds: s0[r] = S^T[kb+8h+r][qc], s1[r] = S^T[kb+16+8h+r][qc]

        // --- within-lane max over 16 scores, one xor-16 to combine halves ---
        float mx = fmaxf(s0[0], s1[0]);
        #pragma unroll
        for (int r = 1; r < 8; ++r) mx = fmaxf(mx, fmaxf(s0[r], s1[r]));
        mx = fmaxf(mx, __shfl_xor(mx, 16, 32));
        const float newm = fmaxf(mr, mx);
        const float sc = __expf(mr - newm);

        // --- exponentiate, pack pairs to f16, row sum ---
        float pk0[4], pk1[4];
        float rs = 0.f;
        #pragma unroll
        for (int j = 0; j < 4; ++j) {
            const float a0 = __expf(s0[2 * j] - newm);
            const float b0 = __expf(s0[2 * j + 1] - newm);
            const float a1 = __expf(s1[2 * j] - newm);
            const float b1 = __expf(s1[2 * j + 1] - newm);
            rs += (a0 + b0) + (a1 + b1);
            pk0[j] = pack2h(a0, b0);   // keys (8h+2j, 8h+2j+1)
            pk1[j] = pack2h(a1, b1);   // keys (16+8h+2j, 16+8h+2j+1)
        }
        rs += __shfl_xor(rs, 16, 32);
        lr = lr * sc + rs;
        mr = newm;

        // --- assemble P^T B-operand: lane(qc,h) needs keys 16h..16h+15 ---
        // own half: h==0 -> pk0 (keys 0-7); h==1 -> pk1 (keys 24-31)
        // exchange the other tile's regs with lane^16
        union { float f[8]; v16h v; } bb;
        #pragma unroll
        for (int j = 0; j < 4; ++j) {
            const float send = h ? pk0[j] : pk1[j];
            const float recv = __shfl_xor(send, 16, 32);
            bb.f[j]     = h ? recv   : pk0[j];   // keys 16h + 2j, 2j+1
            bb.f[j + 4] = h ? pk1[j] : recv;     // keys 16h + 8 + 2j, 2j+1
        }

        // --- rescale accumulators, Y^T += V^T . P^T ---
        #pragma unroll
        for (int r = 0; r < 8; ++r) { acc0[r] *= sc; acc1[r] *= sc; }
        const _Float16* v0 = Vbq + (size_t)qc * NTOK + kb;          // ch = qc
        const _Float16* v1 = Vbq + (size_t)(16 + qc) * NTOK + kb;   // ch = 16+qc
        const v16h va0 = ld16(v0 + 16 * h, v0 + 16 * h + 8);
        const v16h va1 = ld16(v1 + 16 * h, v1 + 16 * h + 8);
        acc0 = __builtin_amdgcn_wmma_f32_16x16x32_f16(false, va0, false, bb.v,
                                                      (short)0, acc0, false, false);
        acc1 = __builtin_amdgcn_wmma_f32_16x16x32_f16(false, va1, false, bb.v,
                                                      (short)0, acc1, false, false);
    }

    // --- finalize: Y[n][o] = acc/l ; lane writes 2x8 consecutive floats ---
    float* Yb = Y + ((size_t)bq * NTOK + qbase + qc) * CI;
    const float inv = 1.0f / lr;
    #pragma unroll
    for (int r = 0; r < 8; ++r) Yb[8 * h + r]      = acc0[r] * inv;
    #pragma unroll
    for (int r = 0; r < 8; ++r) Yb[16 + 8 * h + r] = acc1[r] * inv;
}

// ---------------------------------------------------------------------------
// Kernel 3: wy = W.y + b, per-(q,c) BatchNorm over (B, 64, 64), residual.
// grid: 4q * 64c = 256 blocks, 256 threads.
// ---------------------------------------------------------------------------
__global__ void __launch_bounds__(256)
outproj_kernel(const float* __restrict__ Y, const float* __restrict__ x,
               const float* __restrict__ w_w, const float* __restrict__ w_b,
               const float* __restrict__ gamma, const float* __restrict__ beta,
               float* __restrict__ out)
{
    const int q = blockIdx.x >> 6;
    const int c = blockIdx.x & 63;
    const int qy = q >> 1, qx = q & 1;
    const int tid = threadIdx.x;

    __shared__ float wy[2 * NTOK];      // 32 KB
    __shared__ float rs[8], rs2[8];

    float wrow[CI];
    #pragma unroll
    for (int o = 0; o < CI; ++o) wrow[o] = w_w[c * CI + o];
    const float wb = w_b[c];

    float s = 0.f, s2 = 0.f;
    for (int b = 0; b < 2; ++b) {
        const float* Yb = Y + (size_t)(q * 2 + b) * NTOK * CI;
        for (int n = tid; n < NTOK; n += 256) {
            float a = wb;
            #pragma unroll
            for (int o = 0; o < CI; ++o) a += wrow[o] * Yb[(size_t)n * CI + o];
            wy[b * NTOK + n] = a;
            s += a; s2 += a * a;
        }
    }
    // block reduction for mean / var
    #pragma unroll
    for (int off = 16; off; off >>= 1) {
        s  += __shfl_xor(s,  off, 32);
        s2 += __shfl_xor(s2, off, 32);
    }
    if ((tid & 31) == 0) { rs[tid >> 5] = s; rs2[tid >> 5] = s2; }
    __syncthreads();
    float ts = 0.f, ts2 = 0.f;
    #pragma unroll
    for (int w = 0; w < 8; ++w) { ts += rs[w]; ts2 += rs2[w]; }
    const float mean = ts * (1.f / 8192.f);
    const float var  = ts2 * (1.f / 8192.f) - mean * mean;
    const float g    = gamma[c] * rsqrtf(var + 1e-5f);
    const float bet  = beta[c] - mean * g;

    for (int b = 0; b < 2; ++b) {
        for (int n = tid; n < NTOK; n += 256) {
            const int i = n >> 6, j = n & 63;
            const size_t xi = (((size_t)b * CCH + c) * 128 + qy * 64 + i) * 128
                              + qx * 64 + j;
            out[xi] = wy[b * NTOK + n] * g + bet + x[xi];
        }
    }
}

// ---------------------------------------------------------------------------
extern "C" void kernel_launch(void* const* d_in, const int* in_sizes, int n_in,
                              void* d_out, int out_size, void* d_ws, size_t ws_size,
                              hipStream_t stream) {
    (void)in_sizes; (void)n_in; (void)out_size; (void)ws_size;
    const float* x     = (const float*)d_in[0];
    const float* g_w   = (const float*)d_in[1];
    const float* g_b   = (const float*)d_in[2];
    const float* th_w  = (const float*)d_in[3];
    const float* th_b  = (const float*)d_in[4];
    const float* ph_w  = (const float*)d_in[5];
    const float* ph_b  = (const float*)d_in[6];
    const float* w_w   = (const float*)d_in[7];
    const float* w_b   = (const float*)d_in[8];
    const float* gamma = (const float*)d_in[9];
    const float* beta  = (const float*)d_in[10];
    float* out = (float*)d_out;

    // workspace layout: Q (2MB f16) | K (2MB f16) | Vt (2MB f16) | Y (4MB f32)
    _Float16* Q  = (_Float16*)d_ws;
    _Float16* K  = Q + (size_t)8 * NTOK * CI;
    _Float16* Vt = K + (size_t)8 * NTOK * CI;
    float*    Y  = (float*)(Vt + (size_t)8 * CI * NTOK);

    proj_kernel<<<dim3(512), 256, 0, stream>>>(x, g_w, g_b, th_w, th_b,
                                               ph_w, ph_b, Q, K, Vt);
    flash_kernel<<<dim3(32, 8), 256, 0, stream>>>(Q, K, Vt, Y);
    outproj_kernel<<<dim3(256), 256, 0, stream>>>(Y, x, w_w, w_b, gamma, beta, out);
}
